// Attn_77970836292156
// MI455X (gfx1250) — matrix operand order repaired and checked by reading, not compiled, not run on hardware
//
#include <hip/hip_runtime.h>
#include <math.h>

typedef __attribute__((ext_vector_type(16))) __bf16 v16bf;
typedef __attribute__((ext_vector_type(8)))  __bf16 v8bf;
typedef __attribute__((ext_vector_type(8)))  float  v8f;

#define WMMA_BF16(a, b, c) \
  __builtin_amdgcn_wmma_f32_16x16x32_bf16(false, (a), false, (b), (short)0, (c), false, false)

// ---------- bf16 helpers (RNE, bit-level; avoids reliance on __bf16 arithmetic) ----------
__device__ __forceinline__ __bf16 f2bf(float f) {
  unsigned u = __builtin_bit_cast(unsigned, f);
  u += 0x7fffu + ((u >> 16) & 1u);
  unsigned short h = (unsigned short)(u >> 16);
  return __builtin_bit_cast(__bf16, h);
}
__device__ __forceinline__ float bf2f(__bf16 b) {
  unsigned short h = __builtin_bit_cast(unsigned short, b);
  unsigned u = ((unsigned)h) << 16;
  return __builtin_bit_cast(float, u);
}

// ---------- WMMA fragment loaders ----------
// A fragment (contiguous k order used here, same as B): lane<16 -> row=lane, K0..15 contiguous;
// lane>=16 -> same row, K16..31.  Caller passes p already offset by (lane>=16 ? 16 : 0).
__device__ __forceinline__ v16bf ld_frag_a(const __bf16* p) {
  v8bf x = *(const v8bf*)p;
  v8bf y = *(const v8bf*)(p + 8);
  v16bf r;
#pragma unroll
  for (int i = 0; i < 8; ++i) { r[i] = x[i]; r[i + 8] = y[i]; }
  return r;
}
// B fragment (contiguous k order used here, same as A): lane<16 -> col=lane, K0..15 contiguous; lane>=16 -> K16..31.
// Caller passes p already offset by (lane>=16 ? 16 : 0); 32 contiguous bytes per lane.
__device__ __forceinline__ v16bf ld_frag_b(const __bf16* p) {
  v8bf x = *(const v8bf*)p;
  v8bf y = *(const v8bf*)(p + 8);
  v16bf r;
#pragma unroll
  for (int i = 0; i < 8; ++i) { r[i] = x[i]; r[i + 8] = y[i]; }
  return r;
}

// ---------- f32 -> (hi,lo) bf16 split converters ----------
__global__ void split_kernel(const float* __restrict__ src, __bf16* __restrict__ hi,
                             __bf16* __restrict__ lo, int n) {
  int i = blockIdx.x * blockDim.x + threadIdx.x;
  if (i >= n) return;
  float x = src[i];
  __bf16 h = f2bf(x);
  hi[i] = h;
  lo[i] = f2bf(x - bf2f(h));
}

// src is [rows][cols] (rows = K), dst is [cols][rows] so K becomes contiguous per column.
__global__ void split_T_kernel(const float* __restrict__ src, __bf16* __restrict__ hi,
                               __bf16* __restrict__ lo, int rows, int cols) {
  int i = blockIdx.x * blockDim.x + threadIdx.x;
  if (i >= rows * cols) return;
  int r = i / cols;
  int c = i - r * cols;
  float x = src[i];
  __bf16 h = f2bf(x);
  size_t o = (size_t)c * rows + r;
  hi[o] = h;
  lo[o] = f2bf(x - bf2f(h));
}

// ---------- split-bf16 GEMM: C[M][N] = A[M][K] * Bt[N][K]^T (+bias) ----------
// 256 threads = 8 waves; block tile 128x64; wave tile 32x32 (2x2 wmma tiles); K step 32.
// 3-term split per tile: Ahi*Bhi + Ahi*Blo + Alo*Bhi  (near-f32 precision in f32 accum).
__global__ __launch_bounds__(256) void gemm_split_kernel(
    const __bf16* __restrict__ Ahi, const __bf16* __restrict__ Alo,
    const __bf16* __restrict__ Bhi, const __bf16* __restrict__ Blo,
    float* __restrict__ C, const float* __restrict__ bias,
    int Mdim, int Ndim, int K) {
  const int wave = threadIdx.x >> 5;
  const int lane = threadIdx.x & 31;
  const int lr = lane & 15;
  const int lh = lane >> 4;
  const int wm = wave & 3;
  const int wn = wave >> 2;
  const int m0 = blockIdx.x * 128 + wm * 32;
  const int n0 = blockIdx.y * 64 + wn * 32;

  v8f acc[2][2];
#pragma unroll
  for (int mi = 0; mi < 2; ++mi)
#pragma unroll
    for (int ni = 0; ni < 2; ++ni)
#pragma unroll
      for (int i = 0; i < 8; ++i) acc[mi][ni][i] = 0.0f;

  // unroll 2: lets the scheduler rotate two fragment register sets so next-iter
  // global loads don't WAR-stall against in-flight wmma (kills the v_nop runs).
#pragma unroll 2
  for (int k = 0; k < K; k += 32) {
    v16bf ah[2], al[2], bh[2], bl[2];
#pragma unroll
    for (int mi = 0; mi < 2; ++mi) {
      size_t off = (size_t)(m0 + mi * 16 + lr) * K + k + lh * 16;
      ah[mi] = ld_frag_a(Ahi + off);
      al[mi] = ld_frag_a(Alo + off);
    }
#pragma unroll
    for (int ni = 0; ni < 2; ++ni) {
      size_t off = (size_t)(n0 + ni * 16 + lr) * K + k + lh * 16;
      bh[ni] = ld_frag_b(Bhi + off);
      bl[ni] = ld_frag_b(Blo + off);
    }
    // order products so back-to-back wmma hit distinct accumulators (hides RAW hazard)
#pragma unroll
    for (int mi = 0; mi < 2; ++mi)
#pragma unroll
      for (int ni = 0; ni < 2; ++ni) acc[mi][ni] = WMMA_BF16(ah[mi], bh[ni], acc[mi][ni]);
#pragma unroll
    for (int mi = 0; mi < 2; ++mi)
#pragma unroll
      for (int ni = 0; ni < 2; ++ni) acc[mi][ni] = WMMA_BF16(ah[mi], bl[ni], acc[mi][ni]);
#pragma unroll
    for (int mi = 0; mi < 2; ++mi)
#pragma unroll
      for (int ni = 0; ni < 2; ++ni) acc[mi][ni] = WMMA_BF16(al[mi], bh[ni], acc[mi][ni]);
  }

  // C layout: v8f elem i -> row m0+tile+i (lanes 0-15) / +8 (lanes 16-31), col = lane&15
#pragma unroll
  for (int mi = 0; mi < 2; ++mi)
#pragma unroll
    for (int ni = 0; ni < 2; ++ni) {
      int col = n0 + ni * 16 + lr;
      float bv = bias ? bias[col] : 0.0f;
#pragma unroll
      for (int i = 0; i < 8; ++i) {
        int row = m0 + mi * 16 + i + lh * 8;
        C[(size_t)row * Ndim + col] = acc[mi][ni][i] + bv;
      }
    }
}

// ---------- RMSNorm + RoPE epilogues (one wave per (token, head); dh = 64) ----------
__device__ __forceinline__ void norm_rope_store(const float* __restrict__ src,
                                                const float* __restrict__ g, int lane,
                                                float pos, __bf16* __restrict__ dst) {
  float e1 = src[lane];
  float e2 = src[lane + 32];
  float ss = e1 * e1 + e2 * e2;
#pragma unroll
  for (int m = 16; m >= 1; m >>= 1) ss += __shfl_xor(ss, m, 32);
  float r = rsqrtf(ss * (1.0f / 64.0f) + 1e-6f);
  float n1 = e1 * r * g[lane];
  float n2 = e2 * r * g[lane + 32];
  // inv_freq = 10000^(-lane/32) ; pair (i, i+32) rotated by angle pos*inv
  float inv = __expf(-(float)lane * 0.28782313662425574f);  // ln(1e4)/32
  float s, c;
  sincosf(pos * inv, &s, &c);
  dst[lane]      = f2bf(n1 * c - n2 * s);
  dst[lane + 32] = f2bf(n2 * c + n1 * s);
}

__global__ __launch_bounds__(256) void qkv_epilogue_kernel(
    const float* __restrict__ qkv, const float* __restrict__ gq, const float* __restrict__ gk,
    __bf16* __restrict__ Qb, __bf16* __restrict__ Kb, __bf16* __restrict__ Vt, int N, int L) {
  int gw = (int)((blockIdx.x * blockDim.x + threadIdx.x) >> 5);
  int lane = threadIdx.x & 31;
  int t = gw >> 4;   // token
  int h = gw & 15;   // head
  if (t >= N) return;
  const float* row = qkv + (size_t)t * 3072;
  norm_rope_store(row + h * 64, gq, lane, (float)t, Qb + ((size_t)h * N + t) * 64);
  norm_rope_store(row + 1024 + h * 64, gk, lane, (float)t, Kb + ((size_t)h * L + t) * 64);
  const float* vp = row + 2048 + h * 64;
  __bf16* vdst = Vt + (size_t)h * 64 * L + t;  // V stored transposed [H][dh][L]
  vdst[(size_t)lane * L]        = f2bf(vp[lane]);
  vdst[(size_t)(lane + 32) * L] = f2bf(vp[lane + 32]);
}

__global__ __launch_bounds__(256) void ckv_epilogue_kernel(
    const float* __restrict__ ckv, const float* __restrict__ gck,
    __bf16* __restrict__ Kb, __bf16* __restrict__ Vt, int N, int Mr, int L) {
  int gw = (int)((blockIdx.x * blockDim.x + threadIdx.x) >> 5);
  int lane = threadIdx.x & 31;
  int t = gw >> 4;
  int h = gw & 15;
  if (t >= Mr) return;
  const float* row = ckv + (size_t)t * 2048;
  // concat happens BEFORE rope in the reference -> c_k gets positions N..N+M-1
  norm_rope_store(row + h * 64, gck, lane, (float)(N + t), Kb + ((size_t)h * L + (N + t)) * 64);
  const float* vp = row + 1024 + h * 64;
  __bf16* vdst = Vt + (size_t)h * 64 * L + (N + t);
  vdst[(size_t)lane * L]        = f2bf(vp[lane]);
  vdst[(size_t)(lane + 32) * L] = f2bf(vp[lane + 32]);
}

// ---------- flash attention: 8 waves/block, wave = 16 q-rows, chunks of 64 keys ----------
// 64-key chunks: 16 wmma per iteration vs per-chunk softmax bookkeeping (reductions,
// accumulator rescale) that is fixed-cost -> halves the VALU overhead of 32-key chunks.
__global__ __launch_bounds__(256) void attn_kernel(
    const __bf16* __restrict__ Qb, const __bf16* __restrict__ Kb,
    const __bf16* __restrict__ Vt, float* __restrict__ O, int N, int L, int D) {
  __shared__ __bf16 psh[8 * 16 * 64];  // per-wave 16x64 P transpose staging (2KB/wave)
  const int wave = threadIdx.x >> 5;
  const int lane = threadIdx.x & 31;
  const int lr = lane & 15;
  const int lh = lane >> 4;
  const int h  = blockIdx.y;
  const int q0 = blockIdx.x * 128 + wave * 16;

  const __bf16* Qh = Qb + (size_t)h * N * 64;
  const __bf16* Kh = Kb + (size_t)h * L * 64;
  const __bf16* Vh = Vt + (size_t)h * 64 * L;
  __bf16* Pw = psh + wave * (16 * 64);

  v16bf qf[2];
#pragma unroll
  for (int s = 0; s < 2; ++s)
    qf[s] = ld_frag_a(Qh + (size_t)(q0 + lr) * 64 + s * 32 + lh * 16);

  v8f acc[4];
  float mrow[8], lrow[8];
#pragma unroll
  for (int t = 0; t < 4; ++t)
#pragma unroll
    for (int i = 0; i < 8; ++i) acc[t][i] = 0.0f;
#pragma unroll
  for (int i = 0; i < 8; ++i) { mrow[i] = -1e30f; lrow[i] = 0.0f; }

  for (int kc = 0; kc < L; kc += 64) {
    // S = Q * K^T over dh=64 (two k-steps), four 16-key tiles; scale 1/sqrt(64)
    float sc[4][8];
#pragma unroll
    for (int kt = 0; kt < 4; ++kt) {
      v8f s;
#pragma unroll
      for (int i = 0; i < 8; ++i) s[i] = 0.0f;
#pragma unroll
      for (int ks = 0; ks < 2; ++ks) {
        v16bf bf = ld_frag_b(Kh + (size_t)(kc + kt * 16 + lr) * 64 + ks * 32 + lh * 16);
        s = WMMA_BF16(qf[ks], bf, s);
      }
#pragma unroll
      for (int i = 0; i < 8; ++i) sc[kt][i] = s[i] * 0.125f;
    }
    // online softmax: row stats reduced over the 16 lanes of each half
    float mc[8];
#pragma unroll
    for (int i = 0; i < 8; ++i)
      mc[i] = fmaxf(fmaxf(sc[0][i], sc[1][i]), fmaxf(sc[2][i], sc[3][i]));
#pragma unroll
    for (int m = 8; m >= 1; m >>= 1)
#pragma unroll
      for (int i = 0; i < 8; ++i) mc[i] = fmaxf(mc[i], __shfl_xor(mc[i], m, 32));
    float rs[8];
#pragma unroll
    for (int i = 0; i < 8; ++i) {
      float mn = fmaxf(mrow[i], mc[i]);
      float cf = __expf(mrow[i] - mn);
      rs[i] = 0.0f;
#pragma unroll
      for (int kt = 0; kt < 4; ++kt) {
        float p = __expf(sc[kt][i] - mn);
        Pw[(i + lh * 8) * 64 + kt * 16 + lr] = f2bf(p);  // P row-major [16][64] in LDS
        rs[i] += p;
      }
      mrow[i] = mn;
      lrow[i] *= cf;
#pragma unroll
      for (int t = 0; t < 4; ++t) acc[t][i] *= cf;
    }
#pragma unroll
    for (int m = 8; m >= 1; m >>= 1)
#pragma unroll
      for (int i = 0; i < 8; ++i) rs[i] += __shfl_xor(rs[i], m, 32);
#pragma unroll
    for (int i = 0; i < 8; ++i) lrow[i] += rs[i];

    // drain in-wave LDS stores before cross-lane re-read (CDNA5 split counter)
    asm volatile("s_wait_dscnt 0x0" ::: "memory");
#pragma unroll
    for (int ks = 0; ks < 2; ++ks) {
      v16bf pf = ld_frag_a(Pw + lr * 64 + ks * 32 + lh * 16);
#pragma unroll
      for (int t = 0; t < 4; ++t) {
        v16bf vf = ld_frag_b(Vh + (size_t)(t * 16 + lr) * L + kc + ks * 32 + lh * 16);
        acc[t] = WMMA_BF16(pf, vf, acc[t]);
      }
    }
  }

#pragma unroll
  for (int i = 0; i < 8; ++i) lrow[i] = 1.0f / lrow[i];
#pragma unroll
  for (int t = 0; t < 4; ++t)
#pragma unroll
    for (int i = 0; i < 8; ++i) {
      int row = q0 + i + lh * 8;
      int col = h * 64 + t * 16 + lr;
      O[(size_t)row * D + col] = acc[t][i] * lrow[i];
    }
}

// ---------- host-side launch ----------
extern "C" void kernel_launch(void* const* d_in, const int* in_sizes, int n_in,
                              void* d_out, int out_size, void* d_ws, size_t ws_size,
                              hipStream_t stream) {
  (void)in_sizes; (void)n_in; (void)out_size; (void)ws_size;
  const float* x    = (const float*)d_in[0];
  const float* y    = (const float*)d_in[1];
  const float* Wqkv = (const float*)d_in[2];
  const float* Wckv = (const float*)d_in[3];
  const float* Wout = (const float*)d_in[4];
  const float* bout = (const float*)d_in[5];
  const float* gq   = (const float*)d_in[6];
  const float* gk   = (const float*)d_in[7];
  const float* gck  = (const float*)d_in[8];

  const int N = 2048, Mr = 2048, D = 1024, H = 16, L = 4096;

  char* w = (char*)d_ws;
  size_t off = 0;
  auto take = [&](size_t bytes) -> void* {
    void* p = (void*)(w + off);
    off += (bytes + 255) & ~(size_t)255;
    return p;
  };
  float*  qkv  = (float*) take((size_t)N * 3 * D * 4);
  float*  ckv  = (float*) take((size_t)Mr * 2 * D * 4);
  float*  Oacc = (float*) take((size_t)N * D * 4);
  __bf16* Qb   = (__bf16*)take((size_t)H * N * 64 * 2);
  __bf16* Kb   = (__bf16*)take((size_t)H * L * 64 * 2);
  __bf16* Vt   = (__bf16*)take((size_t)H * 64 * L * 2);
  __bf16* xhi  = (__bf16*)take((size_t)N * D * 2);
  __bf16* xlo  = (__bf16*)take((size_t)N * D * 2);
  __bf16* yhi  = (__bf16*)take((size_t)Mr * D * 2);
  __bf16* ylo  = (__bf16*)take((size_t)Mr * D * 2);
  __bf16* ohi  = (__bf16*)take((size_t)N * D * 2);
  __bf16* olo  = (__bf16*)take((size_t)N * D * 2);
  __bf16* wqh  = (__bf16*)take((size_t)D * 3 * D * 2);
  __bf16* wql  = (__bf16*)take((size_t)D * 3 * D * 2);
  __bf16* wch  = (__bf16*)take((size_t)D * 2 * D * 2);
  __bf16* wcl  = (__bf16*)take((size_t)D * 2 * D * 2);
  __bf16* woh  = (__bf16*)take((size_t)D * D * 2);
  __bf16* wol  = (__bf16*)take((size_t)D * D * 2);

  // 1) f32 -> hi/lo bf16 splits (weights transposed so K is contiguous per out-column)
  int n;
  n = N * D;      split_kernel<<<(n + 255) / 256, 256, 0, stream>>>(x, xhi, xlo, n);
  n = Mr * D;     split_kernel<<<(n + 255) / 256, 256, 0, stream>>>(y, yhi, ylo, n);
  n = D * 3 * D;  split_T_kernel<<<(n + 255) / 256, 256, 0, stream>>>(Wqkv, wqh, wql, D, 3 * D);
  n = D * 2 * D;  split_T_kernel<<<(n + 255) / 256, 256, 0, stream>>>(Wckv, wch, wcl, D, 2 * D);
  n = D * D;      split_T_kernel<<<(n + 255) / 256, 256, 0, stream>>>(Wout, woh, wol, D, D);

  // 2) projection GEMMs (split-bf16 WMMA, f32 accumulate)
  gemm_split_kernel<<<dim3(N / 128, (3 * D) / 64), 256, 0, stream>>>(
      xhi, xlo, wqh, wql, qkv, nullptr, N, 3 * D, D);
  gemm_split_kernel<<<dim3(Mr / 128, (2 * D) / 64), 256, 0, stream>>>(
      yhi, ylo, wch, wcl, ckv, nullptr, Mr, 2 * D, D);

  // 3) RMSNorm + RoPE + bf16 pack (K head-major, V transposed for WMMA B-frags)
  qkv_epilogue_kernel<<<(N * H) / 8, 256, 0, stream>>>(qkv, gq, gk, Qb, Kb, Vt, N, L);
  ckv_epilogue_kernel<<<(Mr * H) / 8, 256, 0, stream>>>(ckv, gck, Kb, Vt, N, Mr, L);

  // 4) flash attention over 4096 keys
  attn_kernel<<<dim3(N / 128, H), 256, 0, stream>>>(Qb, Kb, Vt, Oacc, N, L, D);

  // 5) output projection with bias
  n = N * D;      split_kernel<<<(n + 255) / 256, 256, 0, stream>>>(Oacc, ohi, olo, n);
  gemm_split_kernel<<<dim3(N / 128, D / 64), 256, 0, stream>>>(
      ohi, olo, woh, wol, (float*)d_out, bout, N, D, D);
}
